// attGTN_66065186947192
// MI455X (gfx1250) — matrix operand order repaired and checked by reading, compiled, not run on hardware
//
#include <hip/hip_runtime.h>

#define N_NODES 50000
#define N_EDGES 800000
#define HID     64
#define HEADS   4
#define CONVS   2
#define CLS     16
#define SLOPE   0.2f

typedef __attribute__((ext_vector_type(16))) _Float16 v16h;
typedef __attribute__((ext_vector_type(8)))  float    v8f;

// ---------- ordered-key float min/max via unsigned atomics ----------
__device__ __forceinline__ unsigned fkey(float f) {
    unsigned b = __float_as_uint(f);
    return (b & 0x80000000u) ? ~b : (b | 0x80000000u);
}
__device__ __forceinline__ float funkey(unsigned k) {
    unsigned b = (k & 0x80000000u) ? (k & 0x7fffffffu) : ~k;
    return __uint_as_float(b);
}
__device__ __forceinline__ float lrelu(float x) { return x > 0.f ? x : SLOPE * x; }

// =====================================================================
// f16-WMMA GEMM, fully specialized at compile time:
//   C[M x NCOLS] = op(A [+A2]) @ W[K x NCOLS] (+bias)
// HASA2: A_val = A + A2;  RELUA: A_val = relu(A_val);  ACC: C += tile.
// One wave computes one 16x16 tile; 8 waves (256 thr) per block.
// A fragment per lane = 4 aligned float4 loads (ISA 7.12.2 A layout).
// Epilogue: wave-uniform full-tile fast path (no per-element guards).
// =====================================================================
template <int K, int NCOLS, int HASA2, int RELUA, int ACC>
__global__ void wmma_gemm(const float* __restrict__ A,
                          const float* __restrict__ A2,
                          int M,
                          const float* __restrict__ W,
                          const float* __restrict__ bias,
                          float* __restrict__ C) {
    constexpr int colTiles = NCOLS >> 4;           // 4 (NCOLS=64) or 1 (NCOLS=16)
    constexpr int rtpb     = 8 / colTiles;         // row tiles per block
    const int lane    = threadIdx.x & 31;
    const int wv      = threadIdx.x >> 5;
    const int rowTile = blockIdx.x * rtpb + (wv / colTiles);
    const int colTile = wv % colTiles;
    if (rowTile * 16 >= M) return;                 // wave-uniform exit

    const int half = lane >> 4;                    // 0 | 1
    const int lrow = lane & 15;
    int r = rowTile * 16 + lrow;  if (r >= M) r = M - 1;   // clamp loads
    const int ncol = colTile * 16 + lrow;
    const int kbA  = half << 3;                    // A: +0 / +8
    const int kbB  = half << 3;                    // B: +0 / +8

    const float* __restrict__ arow  = A  + (long)r * K + kbA;
    const float* __restrict__ arow2 = HASA2 ? (A2 + (long)r * K + kbA) : nullptr;

    v8f c = {};
    for (int kk = 0; kk < K; kk += 32) {
        // ---- A fragment: j=0..7 -> K=kk+kbA+0..7 ; j=8..15 -> +16..23 ----
        float av[16];
        {
            const float4* ap = (const float4*)(arow + kk);
            float4 p0 = ap[0], p1 = ap[1], p2 = ap[4], p3 = ap[5];
            if (HASA2) {
                const float4* ap2 = (const float4*)(arow2 + kk);
                float4 q0 = ap2[0], q1 = ap2[1], q2 = ap2[4], q3 = ap2[5];
                p0.x += q0.x; p0.y += q0.y; p0.z += q0.z; p0.w += q0.w;
                p1.x += q1.x; p1.y += q1.y; p1.z += q1.z; p1.w += q1.w;
                p2.x += q2.x; p2.y += q2.y; p2.z += q2.z; p2.w += q2.w;
                p3.x += q3.x; p3.y += q3.y; p3.z += q3.z; p3.w += q3.w;
            }
            av[0]=p0.x; av[1]=p0.y; av[2]=p0.z; av[3]=p0.w;
            av[4]=p1.x; av[5]=p1.y; av[6]=p1.z; av[7]=p1.w;
            av[8]=p2.x; av[9]=p2.y; av[10]=p2.z; av[11]=p2.w;
            av[12]=p3.x; av[13]=p3.y; av[14]=p3.z; av[15]=p3.w;
        }
        union { v16h v; _Float16 h[16]; } a, b;
#pragma unroll
        for (int j = 0; j < 16; ++j) {
            float v = av[j];
            if (RELUA) v = fmaxf(v, 0.f);
            a.h[j] = (_Float16)v;
            // ---- B fragment: j=0..7 -> K=kk+kbB+0..7 ; j=8..15 -> +16..23, column ncol ----
            b.h[j] = (_Float16)W[(long)(kk + kbB + (j & 7) + ((j >> 3) << 4)) * NCOLS + ncol];
        }
        c = __builtin_amdgcn_wmma_f32_16x16x32_f16(
                false, a.v, false, b.v, (short)0, c, false, false);
    }

    union { v8f v; float f[8]; } cu; cu.v = c;
    float* __restrict__ cp = C + (long)(rowTile * 16 + (half << 3)) * NCOLS + ncol;

    if (rowTile * 16 + 16 <= M) {
        // full tile: unguarded straight-line epilogue (common case)
#pragma unroll
        for (int j = 0; j < 8; ++j) {
            long idx = (long)j * NCOLS;
            if (ACC) cp[idx] += cu.f[j];
            else     cp[idx]  = cu.f[j] + bias[ncol];
        }
    } else {
        // ragged last tile only
#pragma unroll
        for (int j = 0; j < 8; ++j) {
            int row = rowTile * 16 + j + (half << 3);
            if (row < M) {
                long idx = (long)j * NCOLS;
                if (ACC) cp[idx] += cu.f[j];
                else     cp[idx]  = cu.f[j] + bias[ncol];
            }
        }
    }
}

// =====================================================================
// Per-node row sum of x[N,64] -> s[n]; init min/max keys. 1 wave / node.
// =====================================================================
__global__ void row_sum_init(const float* __restrict__ x, float* __restrict__ s,
                             unsigned* __restrict__ smaxk, unsigned* __restrict__ smink,
                             int N) {
    int wid  = (blockIdx.x * blockDim.x + threadIdx.x) >> 5;
    int lane = threadIdx.x & 31;
    if (wid >= N) return;
    const float2* row = (const float2*)(x + (long)wid * HID);
    float2 v = row[lane];
    float sum = v.x + v.y;
#pragma unroll
    for (int off = 16; off; off >>= 1) sum += __shfl_xor(sum, off, 32);
    if (lane == 0) { s[wid] = sum; smaxk[wid] = 0u; smink[wid] = 0xFFFFFFFFu; }
}

// =====================================================================
// Per-edge scalar min/max of s[src] into dst buckets (2 atomics / edge).
// =====================================================================
__global__ void edge_minmax(const int* __restrict__ src, const int* __restrict__ dst,
                            const float* __restrict__ s,
                            unsigned* __restrict__ smaxk, unsigned* __restrict__ smink,
                            int E) {
    int e = blockIdx.x * blockDim.x + threadIdx.x;
    if (e >= E) return;
    unsigned key = fkey(s[src[e]]);
    int v = dst[e];
    atomicMax(&smaxk[v], key);
    atomicMin(&smink[v], key);
}

// =====================================================================
// Per-(node,k): softmax max  m = lrelu(al*(al>=0?smax:smin) + ar*s_dst)
// (lrelu is monotone, so segment-max collapses to scalar min/max of s_src)
// Also zeroes den/acc accumulators.
// =====================================================================
__global__ void node_prep(const float* __restrict__ s,
                          const unsigned* __restrict__ smaxk,
                          const unsigned* __restrict__ smink,
                          const float* __restrict__ al, const float* __restrict__ ar,
                          float* __restrict__ mbuf, float* __restrict__ den,
                          float* __restrict__ acc, int N) {
    int t = blockIdx.x * blockDim.x + threadIdx.x;
    int n = t >> 6, k = t & 63;
    if (n >= N) return;
    float a  = al[k];
    float sb = (a >= 0.f) ? funkey(smaxk[n]) : funkey(smink[n]);
    mbuf[t] = lrelu(a * sb + ar[k] * s[n]);
    den[t] = 0.f;
    acc[t] = 0.f;
}

// =====================================================================
// Main edge pass: per (edge, 4-channel group):
//   ex = exp(lrelu(al*s_src + ar*s_dst) - m[dst]);
//   den[dst] += ex;  acc[dst] += x[src]*ex;
// =====================================================================
__global__ void edge_scatter(const int* __restrict__ src, const int* __restrict__ dst,
                             const float* __restrict__ s, const float* __restrict__ mbuf,
                             const float* __restrict__ x,
                             const float* __restrict__ al, const float* __restrict__ ar,
                             float* __restrict__ den, float* __restrict__ acc, int E) {
    int t = blockIdx.x * blockDim.x + threadIdx.x;
    int e  = t >> 4;
    int k0 = (t & 15) << 2;
    if (e >= E) return;
    int u = src[e], v = dst[e];
    float su = s[u], sv = s[v];
    long vb = (long)v * HID + k0;
    long ub = (long)u * HID + k0;
    float4 alv = *(const float4*)(al + k0);
    float4 arv = *(const float4*)(ar + k0);
    float4 mv  = *(const float4*)(mbuf + vb);
    float4 xv  = *(const float4*)(x + ub);
    float ex0 = __expf(lrelu(alv.x * su + arv.x * sv) - mv.x);
    float ex1 = __expf(lrelu(alv.y * su + arv.y * sv) - mv.y);
    float ex2 = __expf(lrelu(alv.z * su + arv.z * sv) - mv.z);
    float ex3 = __expf(lrelu(alv.w * su + arv.w * sv) - mv.w);
    atomicAdd(&den[vb + 0], ex0);
    atomicAdd(&den[vb + 1], ex1);
    atomicAdd(&den[vb + 2], ex2);
    atomicAdd(&den[vb + 3], ex3);
    atomicAdd(&acc[vb + 0], xv.x * ex0);
    atomicAdd(&acc[vb + 1], xv.y * ex1);
    atomicAdd(&acc[vb + 2], xv.z * ex2);
    atomicAdd(&acc[vb + 3], xv.w * ex3);
}

// =====================================================================
// xn = acc/den; fused row-sum -> s for next conv; reset min/max keys.
// 1 wave / node (lane handles k and k+32).
// =====================================================================
__global__ void node_div(const float* __restrict__ acc, const float* __restrict__ den,
                         float* __restrict__ xn, float* __restrict__ s,
                         unsigned* __restrict__ smaxk, unsigned* __restrict__ smink,
                         int N) {
    int wid  = (blockIdx.x * blockDim.x + threadIdx.x) >> 5;
    int lane = threadIdx.x & 31;
    if (wid >= N) return;
    long base = (long)wid * HID;
    float a0 = acc[base + lane]      / den[base + lane];
    float a1 = acc[base + lane + 32] / den[base + lane + 32];
    xn[base + lane]      = a0;
    xn[base + lane + 32] = a1;
    float sum = a0 + a1;
#pragma unroll
    for (int off = 16; off; off >>= 1) sum += __shfl_xor(sum, off, 32);
    if (lane == 0) { s[wid] = sum; smaxk[wid] = 0u; smink[wid] = 0xFFFFFFFFu; }
}

// =====================================================================
extern "C" void kernel_launch(void* const* d_in, const int* in_sizes, int n_in,
                              void* d_out, int out_size, void* d_ws, size_t ws_size,
                              hipStream_t stream) {
    const float* feat0 = (const float*)d_in[0];
    const float* feat1 = (const float*)d_in[1];
    const float* feat2 = (const float*)d_in[2];
    const int*   src   = (const int*)d_in[3];
    const int*   dst   = (const int*)d_in[4];
    const float* W1_0  = (const float*)d_in[5];
    const float* b1_0  = (const float*)d_in[6];
    const float* W1_1  = (const float*)d_in[7];
    const float* b1_1  = (const float*)d_in[8];
    const float* W1_2  = (const float*)d_in[9];
    const float* b1_2  = (const float*)d_in[10];
    const float* al    = (const float*)d_in[11];   // [HEADS][CONVS][HID]
    const float* ar    = (const float*)d_in[12];
    const float* W2    = (const float*)d_in[13];   // [256][64]
    const float* b2    = (const float*)d_in[14];
    const float* W3    = (const float*)d_in[15];   // [64][16]
    const float* b3    = (const float*)d_in[16];

    float* out     = (float*)d_out;
    float* logits  = out;                          // [N,16]
    float* enc     = out + (long)N_NODES * CLS;    // [N,64]

    // workspace layout (floats)
    float* ws = (float*)d_ws;
    const long NH = (long)N_NODES * HID;           // 3.2M
    float*    h     = ws;
    float*    xA    = h   + NH;
    float*    xB    = xA  + NH;
    float*    mbuf  = xB  + NH;
    float*    den   = mbuf+ NH;
    float*    acc   = den + NH;
    float*    s     = acc + NH;
    unsigned* smaxk = (unsigned*)(s + N_NODES);
    unsigned* smink = smaxk + N_NODES;

    auto cdiv = [](long a, long b) { return (int)((a + b - 1) / b); };

    // ---- fc1: per-type projection into h[N,64] --------------------------
    wmma_gemm<256, 64, 0, 0, 0><<<cdiv(20000, 32), 256, 0, stream>>>(
        feat0, nullptr, 20000, W1_0, b1_0, h);
    wmma_gemm<128, 64, 0, 0, 0><<<cdiv(15000, 32), 256, 0, stream>>>(
        feat1, nullptr, 15000, W1_1, b1_1, h + 20000L * HID);
    wmma_gemm< 64, 64, 0, 0, 0><<<cdiv(15000, 32), 256, 0, stream>>>(
        feat2, nullptr, 15000, W1_2, b1_2, h + 35000L * HID);

    const int nwaveBlocks = cdiv((long)N_NODES * 32, 256);     // 1 wave / node
    const int edgeBlocks  = cdiv(N_EDGES, 256);
    const int nkBlocks    = cdiv((long)N_NODES * HID, 256);
    const int scatBlocks  = cdiv((long)N_EDGES * 16, 256);

    // ---- heads ----------------------------------------------------------
    for (int hd = 0; hd < HEADS; ++hd) {
        row_sum_init<<<nwaveBlocks, 256, 0, stream>>>(h, s, smaxk, smink, N_NODES);
        const float* x = h;
        for (int c = 0; c < CONVS; ++c) {
            const float* alp = al + (long)(hd * CONVS + c) * HID;
            const float* arp = ar + (long)(hd * CONVS + c) * HID;
            float* xn = (c == 0) ? xA : xB;
            edge_minmax <<<edgeBlocks, 256, 0, stream>>>(src, dst, s, smaxk, smink, N_EDGES);
            node_prep   <<<nkBlocks,   256, 0, stream>>>(s, smaxk, smink, alp, arp,
                                                         mbuf, den, acc, N_NODES);
            edge_scatter<<<scatBlocks, 256, 0, stream>>>(src, dst, s, mbuf, x, alp, arp,
                                                         den, acc, N_EDGES);
            node_div    <<<nwaveBlocks,256, 0, stream>>>(acc, den, xn, s, smaxk, smink, N_NODES);
            x = xn;
        }
        // encoded (+)= relu(h + x_head) @ W2[hd*64 : hd*64+64, :]  (+b2 first)
        if (hd == 0)
            wmma_gemm<64, 64, 1, 1, 0><<<cdiv(N_NODES, 32), 256, 0, stream>>>(
                h, xB, N_NODES, W2 + (long)hd * HID * HID, b2, enc);
        else
            wmma_gemm<64, 64, 1, 1, 1><<<cdiv(N_NODES, 32), 256, 0, stream>>>(
                h, xB, N_NODES, W2 + (long)hd * HID * HID, b2, enc);
    }

    // ---- fc3: logits = relu(encoded) @ W3 + b3 --------------------------
    wmma_gemm<64, 16, 0, 1, 0><<<cdiv(N_NODES, 128), 256, 0, stream>>>(
        enc, nullptr, N_NODES, W3, b3, logits);
}
